// LinearMessagePassingLayer_32109175505236
// MI455X (gfx1250) — matrix operand order repaired and checked by reading, not compiled, not run on hardware
//
#include <hip/hip_runtime.h>
#include <hip/hip_bf16.h>

// ---------------------------------------------------------------------------
// LinearMessagePassingLayer for MI455X (gfx1250), bf16 WMMA + f32 accumulate.
//   msg   = [nodes[senders] || edges] @ W_message          (16x128 @ 128x64)
//   agg   = segment_sum(msg, receivers)                     (atomic f32 adds)
//   h     = relu(relu([nodes||agg]@W0+b0)@W1+b1)@W2+b2
//   out   = LayerNorm(nodes@W_node + h) * scale + bias
// ---------------------------------------------------------------------------

typedef __attribute__((ext_vector_type(8)))  __bf16 v8bf;
typedef __attribute__((ext_vector_type(16))) __bf16 v16bf;
typedef __attribute__((ext_vector_type(8)))  float  v8f;
typedef __attribute__((ext_vector_type(4)))  float  v4f;

#define WMMA_BF16(a, b, c) \
  __builtin_amdgcn_wmma_f32_16x16x32_bf16(false, (a), false, (b), (short)0, (c), false, false)

static __device__ __forceinline__ v16bf mk16(v8bf lo, v8bf hi) {
  return __builtin_shufflevector(lo, hi, 0,1,2,3,4,5,6,7,8,9,10,11,12,13,14,15);
}

// A fragment: lane L holds row M = L&15; half hh = L>>4 selects K sub-groups.
// VGPR pairs: K = k0 + hh*8 + {0..7}  then  K = k0 + 16 + hh*8 + {0..7}.
static __device__ __forceinline__ v16bf a_frag(const __bf16* row, int k0, int hh) {
  v8bf lo = *(const v8bf*)(row + k0 + hh * 8);
  v8bf hi = *(const v8bf*)(row + k0 + 16 + hh * 8);
  return mk16(lo, hi);
}

// B fragment: lane L holds column N = L&15; K = k0 + hh*8 + {0..7} then k0 + 16 + hh*8 + {0..7}
// (Wt stored as [n][k], contiguous in k).
static __device__ __forceinline__ v16bf b_frag(const __bf16* row, int k0, int hh) {
  v8bf lo = *(const v8bf*)(row + k0 + hh * 8);
  v8bf hi = *(const v8bf*)(row + k0 + 16 + hh * 8);
  return mk16(lo, hi);
}

// ---------------------------------------------------------------------------
// Prep: zero aggregation buffer; transpose + convert weights to bf16 Wt[n][k].
// ---------------------------------------------------------------------------
__global__ __launch_bounds__(256) void lmp_prep(
    const float* __restrict__ Wmsg, const float* __restrict__ Wn,
    const float* __restrict__ w0,   const float* __restrict__ w1,
    const float* __restrict__ w2,
    float* __restrict__ agg,
    __bf16* __restrict__ wmsgT, __bf16* __restrict__ w0T,
    __bf16* __restrict__ w1T,   __bf16* __restrict__ w2T,
    __bf16* __restrict__ wnT,   int aggCount) {
  int stride = gridDim.x * blockDim.x;
  int tid = blockIdx.x * blockDim.x + threadIdx.x;
  for (int i = tid; i < aggCount; i += stride) agg[i] = 0.0f;
  for (int i = tid; i < 128 * 64; i += stride) {        // W_message [128,64]
    int k = i >> 6, n = i & 63;
    wmsgT[n * 128 + k] = (__bf16)Wmsg[i];
  }
  for (int i = tid; i < 128 * 128; i += stride) {       // mlp_w0/w1 [128,128]
    int k = i >> 7, n = i & 127;
    w0T[n * 128 + k] = (__bf16)w0[i];
    w1T[n * 128 + k] = (__bf16)w1[i];
  }
  for (int i = tid; i < 128 * 64; i += stride) {        // mlp_w2 [128,64]
    int k = i >> 6, n = i & 63;
    w2T[n * 128 + k] = (__bf16)w2[i];
  }
  for (int i = tid; i < 64 * 64; i += stride) {         // W_node [64,64]
    int k = i >> 6, n = i & 63;
    wnT[n * 64 + k] = (__bf16)Wn[i];
  }
}

// ---------------------------------------------------------------------------
// Messages: one wave per 16-edge tile. A = [sender_feat || edge_feat] (16x128),
// B = W_message (128x64). 4 N-tiles x 4 K-chunks of wmma_f32_16x16x32_bf16,
// then atomic scatter-add of the f32 accumulators into agg[receiver].
// ---------------------------------------------------------------------------
__global__ __launch_bounds__(256) void lmp_messages(
    const float* __restrict__ nodes, const float* __restrict__ edges,
    const int* __restrict__ senders, const int* __restrict__ receivers,
    const __bf16* __restrict__ wmsgT, float* __restrict__ agg, int nEdges) {
  __shared__ __attribute__((aligned(16))) __bf16 sA[8][16 * 128];
  const int wave = threadIdx.x >> 5;
  const int lane = threadIdx.x & 31;
  const int tile = blockIdx.x * 8 + wave;
  const int e0 = tile * 16;
  if (e0 >= nEdges) return;

  __bf16* A = sA[wave];
  {  // stage 16x128 bf16 A tile: each lane does one half-row (64 floats)
    const int m = lane >> 1, half = lane & 1;
    int e = e0 + m; if (e >= nEdges) e = nEdges - 1;
    const float* src = half ? (edges + (size_t)e * 64)
                            : (nodes + (size_t)senders[e] * 64);
    __bf16* dst = A + m * 128 + half * 64;
#pragma unroll
    for (int j = 0; j < 64; j += 4) {
      v4f v = *(const v4f*)(src + j);
      dst[j + 0] = (__bf16)v.x; dst[j + 1] = (__bf16)v.y;
      dst[j + 2] = (__bf16)v.z; dst[j + 3] = (__bf16)v.w;
    }
  }
  __builtin_amdgcn_wave_barrier();

  const int nl = lane & 15, hh = lane >> 4;
  const __bf16* Arow = A + nl * 128;

  int recv[8];
#pragma unroll
  for (int r = 0; r < 8; ++r) {
    int e = e0 + r + 8 * hh; if (e >= nEdges) e = nEdges - 1;
    recv[r] = receivers[e];
  }

#pragma unroll
  for (int t = 0; t < 4; ++t) {
    v8f acc;
#pragma unroll
    for (int r = 0; r < 8; ++r) acc[r] = 0.0f;
    const __bf16* Brow = wmsgT + (size_t)(t * 16 + nl) * 128;
#pragma unroll
    for (int kc = 0; kc < 4; ++kc)
      acc = WMMA_BF16(a_frag(Arow, kc * 32, hh), b_frag(Brow, kc * 32, hh), acc);
    const int col = t * 16 + nl;
#pragma unroll
    for (int r = 0; r < 8; ++r)
      unsafeAtomicAdd(agg + (size_t)recv[r] * 64 + col, acc[r]);
  }
}

// ---------------------------------------------------------------------------
// Node update: one wave per 16-node tile. Three WMMA GEMM layers + residual
// WMMA GEMM, then LayerNorm via __shfl_xor reductions within 16-lane halves.
// ---------------------------------------------------------------------------
__global__ __launch_bounds__(128) void lmp_nodes(
    const float* __restrict__ nodes, const float* __restrict__ agg,
    const __bf16* __restrict__ w0T, const __bf16* __restrict__ w1T,
    const __bf16* __restrict__ w2T, const __bf16* __restrict__ wnT,
    const float* __restrict__ b0, const float* __restrict__ b1,
    const float* __restrict__ b2, const float* __restrict__ lns,
    const float* __restrict__ lnb, float* __restrict__ out, int nNodes) {
  __shared__ __attribute__((aligned(16))) __bf16 sIn[4][16 * 128];
  __shared__ __attribute__((aligned(16))) __bf16 sH[4][16 * 128];
  const int wave = threadIdx.x >> 5;
  const int lane = threadIdx.x & 31;
  const int tile = blockIdx.x * 4 + wave;
  const int n0 = tile * 16;
  if (n0 >= nNodes) return;

  __bf16* In = sIn[wave];
  __bf16* H  = sH[wave];
  {  // stage A = [nodes || agg] (16x128 bf16)
    const int m = lane >> 1, half = lane & 1;
    int nd = n0 + m; if (nd >= nNodes) nd = nNodes - 1;
    const float* src = half ? (agg + (size_t)nd * 64) : (nodes + (size_t)nd * 64);
    __bf16* dst = In + m * 128 + half * 64;
#pragma unroll
    for (int j = 0; j < 64; j += 4) {
      v4f v = *(const v4f*)(src + j);
      dst[j + 0] = (__bf16)v.x; dst[j + 1] = (__bf16)v.y;
      dst[j + 2] = (__bf16)v.z; dst[j + 3] = (__bf16)v.w;
    }
  }
  __builtin_amdgcn_wave_barrier();

  const int nl = lane & 15, hh = lane >> 4;
  const __bf16* ArowIn = In + nl * 128;
  const __bf16* ArowH  = H  + nl * 128;

  // ---- layer 0: relu(In @ W0 + b0) -> H ----
#pragma unroll
  for (int t = 0; t < 8; ++t) {
    const float bv = b0[t * 16 + nl];
    v8f c;
#pragma unroll
    for (int r = 0; r < 8; ++r) c[r] = bv;
    const __bf16* Brow = w0T + (size_t)(t * 16 + nl) * 128;
#pragma unroll
    for (int kc = 0; kc < 4; ++kc)
      c = WMMA_BF16(a_frag(ArowIn, kc * 32, hh), b_frag(Brow, kc * 32, hh), c);
    const int col = t * 16 + nl;
#pragma unroll
    for (int r = 0; r < 8; ++r)
      H[(r + 8 * hh) * 128 + col] = (__bf16)fmaxf(c[r], 0.0f);
  }
  __builtin_amdgcn_wave_barrier();

  // ---- layer 1: relu(H @ W1 + b1) -> H (read all before overwrite) ----
  {
    v8f acc[8];
#pragma unroll
    for (int t = 0; t < 8; ++t) {
      const float bv = b1[t * 16 + nl];
      v8f c;
#pragma unroll
      for (int r = 0; r < 8; ++r) c[r] = bv;
      const __bf16* Brow = w1T + (size_t)(t * 16 + nl) * 128;
#pragma unroll
      for (int kc = 0; kc < 4; ++kc)
        c = WMMA_BF16(a_frag(ArowH, kc * 32, hh), b_frag(Brow, kc * 32, hh), c);
      acc[t] = c;
    }
    __builtin_amdgcn_wave_barrier();
#pragma unroll
    for (int t = 0; t < 8; ++t) {
      const int col = t * 16 + nl;
#pragma unroll
      for (int r = 0; r < 8; ++r)
        H[(r + 8 * hh) * 128 + col] = (__bf16)fmaxf(acc[t][r], 0.0f);
    }
  }
  __builtin_amdgcn_wave_barrier();

  // ---- layer 2 (H @ W2 + b2) + residual (nodes @ W_node) ----
  v8f x[4];
#pragma unroll
  for (int t = 0; t < 4; ++t) {
    const float bv = b2[t * 16 + nl];
    v8f c;
#pragma unroll
    for (int r = 0; r < 8; ++r) c[r] = bv;
    const __bf16* Brow2 = w2T + (size_t)(t * 16 + nl) * 128;
#pragma unroll
    for (int kc = 0; kc < 4; ++kc)
      c = WMMA_BF16(a_frag(ArowH, kc * 32, hh), b_frag(Brow2, kc * 32, hh), c);
    const __bf16* BrowN = wnT + (size_t)(t * 16 + nl) * 64;
#pragma unroll
    for (int kc = 0; kc < 2; ++kc)
      c = WMMA_BF16(a_frag(ArowIn, kc * 32, hh), b_frag(BrowN, kc * 32, hh), c);
    x[t] = c;
  }

  // ---- LayerNorm over 64 features per row ----
  float lnsv[4], lnbv[4];
#pragma unroll
  for (int t = 0; t < 4; ++t) { lnsv[t] = lns[t * 16 + nl]; lnbv[t] = lnb[t * 16 + nl]; }

#pragma unroll
  for (int r = 0; r < 8; ++r) {
    float s = 0.0f, s2 = 0.0f;
#pragma unroll
    for (int t = 0; t < 4; ++t) { float v = x[t][r]; s += v; s2 += v * v; }
#pragma unroll
    for (int mask = 1; mask < 16; mask <<= 1) {
      s  += __shfl_xor(s,  mask);
      s2 += __shfl_xor(s2, mask);
    }
    const float mu  = s * (1.0f / 64.0f);
    const float var = s2 * (1.0f / 64.0f) - mu * mu;
    const float inv = rsqrtf(var + 1e-6f);
    int row = n0 + r + 8 * hh; if (row >= nNodes) row = nNodes - 1;
#pragma unroll
    for (int t = 0; t < 4; ++t) {
      const float v = (x[t][r] - mu) * inv * lnsv[t] + lnbv[t];
      out[(size_t)row * 64 + t * 16 + nl] = v;
    }
  }
}

// ---------------------------------------------------------------------------
extern "C" void kernel_launch(void* const* d_in, const int* in_sizes, int n_in,
                              void* d_out, int out_size, void* d_ws, size_t ws_size,
                              hipStream_t stream) {
  const float* nodes     = (const float*)d_in[0];
  const float* edges     = (const float*)d_in[1];
  const int*   senders   = (const int*)d_in[2];
  const int*   receivers = (const int*)d_in[3];
  const float* Wmsg      = (const float*)d_in[4];
  const float* Wn        = (const float*)d_in[5];
  const float* w0        = (const float*)d_in[6];
  const float* b0        = (const float*)d_in[7];
  const float* w1        = (const float*)d_in[8];
  const float* b1        = (const float*)d_in[9];
  const float* w2        = (const float*)d_in[10];
  const float* b2        = (const float*)d_in[11];
  const float* lns       = (const float*)d_in[12];
  const float* lnb       = (const float*)d_in[13];
  float* out = (float*)d_out;

  const int N = in_sizes[0] / 64;
  const int E = in_sizes[2];

  char* ws = (char*)d_ws;
  float* agg = (float*)ws;
  size_t off = (((size_t)N * 64 * sizeof(float)) + 255) & ~(size_t)255;
  __bf16* wmsgT = (__bf16*)(ws + off); off += (size_t)64  * 128 * 2;
  __bf16* w0T   = (__bf16*)(ws + off); off += (size_t)128 * 128 * 2;
  __bf16* w1T   = (__bf16*)(ws + off); off += (size_t)128 * 128 * 2;
  __bf16* w2T   = (__bf16*)(ws + off); off += (size_t)64  * 128 * 2;
  __bf16* wnT   = (__bf16*)(ws + off); off += (size_t)64  * 64  * 2;

  lmp_prep<<<512, 256, 0, stream>>>(Wmsg, Wn, w0, w1, w2, agg,
                                    wmsgT, w0T, w1T, w2T, wnT, N * 64);

  const int tilesE = (E + 15) / 16;
  lmp_messages<<<(tilesE + 7) / 8, 256, 0, stream>>>(
      nodes, edges, senders, receivers, wmsgT, agg, E);

  const int tilesN = (N + 15) / 16;
  lmp_nodes<<<(tilesN + 3) / 4, 128, 0, stream>>>(
      nodes, agg, w0T, w1T, w2T, wnT, b0, b1, b2, lns, lnb, out, N);
}
